// ResidualVectorQuantizer_77910706749688
// MI455X (gfx1250) — matrix operand order repaired and checked by reading, not compiled, not run on hardware
//
#include <hip/hip_runtime.h>
#include <hip/hip_bf16.h>

typedef __attribute__((ext_vector_type(16))) __bf16 v16bf;
typedef __attribute__((ext_vector_type(8)))  __bf16 v8bf;
typedef __attribute__((ext_vector_type(8)))  float  v8f;

static constexpr int kN    = 8192;  // rows of x
static constexpr int kDim  = 512;   // feature dim
static constexpr int kEmb  = 8192;  // codebook size
static constexpr int kChunks    = 4;                  // codebook split for occupancy
static constexpr int kChunkSize = kEmb / kChunks;     // 2048 codes per chunk

// Block geometry for the distance kernel: 8 waves x 16 rows = 128 rows/block.
static constexpr int kWavesPerBlock = 8;
static constexpr int kBlockThreads  = kWavesPerBlock * 32;

// LDS staging geometry for one 16-row x 512-col bf16 B tile, padded to dodge
// bank conflicts: row stride 520 bf16 = 1040 B = 260 dwords (stride % 64 = 4).
static constexpr int kLdsRow   = 520;                 // bf16 elements per row
static constexpr int kLdsTileB = 16 * kLdsRow * 2;    // bytes per buffer (16640)

// ---------------------------------------------------------------------------
// CDNA5 async global->LDS copy (ASYNCcnt-tracked), per CDNA5 ISA
// ---------------------------------------------------------------------------
__device__ __forceinline__ void async_load_b128(const void* gptr, unsigned ldsByteOff) {
  asm volatile("global_load_async_to_lds_b128 %0, %1, off"
               :: "v"(ldsByteOff), "v"(gptr)
               : "memory");
}
__device__ __forceinline__ void wait_asynccnt0() {
  asm volatile("s_wait_asynccnt 0x0" ::: "memory");
}

// ---------------------------------------------------------------------------
// Kernel 1: generic f32 -> bf16 convert
// ---------------------------------------------------------------------------
__global__ void cvt_bf16_kernel(const float* __restrict__ in,
                                __hip_bfloat16* __restrict__ out, int n) {
  int i = blockIdx.x * blockDim.x + threadIdx.x;
  if (i < n) out[i] = __float2bfloat16(in[i]);
}

// ---------------------------------------------------------------------------
// Kernel 2: codebook row norms + bf16 convert (one wave per codebook row)
// ---------------------------------------------------------------------------
__global__ void enorm_cvt_kernel(const float* __restrict__ E,
                                 __hip_bfloat16* __restrict__ eb,
                                 float* __restrict__ enorm) {
  int lane = threadIdx.x & 31;
  int wave = threadIdx.x >> 5;
  int row  = blockIdx.x * (blockDim.x >> 5) + wave;
  if (row >= kEmb) return;
  const float* p = E + (size_t)row * kDim;
  __hip_bfloat16* q = eb + (size_t)row * kDim;
  float s = 0.f;
#pragma unroll
  for (int i = 0; i < kDim / 32; ++i) {
    float v = p[lane + 32 * i];
    s += v * v;
    q[lane + 32 * i] = __float2bfloat16(v);
  }
#pragma unroll
  for (int m = 16; m >= 1; m >>= 1) s += __shfl_xor(s, m, 32);
  if (lane == 0) enorm[row] = s;
}

// ---------------------------------------------------------------------------
// Kernel 3: fused distance GEMM + partial argmin over one codebook chunk.
// Block = 8 waves x 16 rows of x. A (16x512 bf16 per wave) resident in VGPRs.
// B tiles (16 codebook rows, 16 KB) staged into LDS once per block with
// double-buffered global_load_async_to_lds_b128 and consumed via ds_load_b128
// by all 8 waves -> 8x less L2 traffic than per-wave global streaming.
// B fragments rotate through 4 register buffers and a sched_group_barrier
// pipeline (8 DS reads, then {WMMA, 2 DS reads}x12, then 4 WMMAs) keeps ~3
// WMMA-durations of LDS latency in flight per wave.
// ---------------------------------------------------------------------------
__global__ __launch_bounds__(kBlockThreads) void dist_argmin_kernel(
    const __hip_bfloat16* __restrict__ xb_,
    const __hip_bfloat16* __restrict__ eb_,
    const float* __restrict__ enorm,
    float* __restrict__ pbestv,
    int* __restrict__ pbesti) {
  const __bf16* xb = (const __bf16*)xb_;
  const __bf16* eb = (const __bf16*)eb_;

  __shared__ __align__(16) __bf16 smem[2][16][kLdsRow];

  int tid  = threadIdx.x;
  int lane = tid & 31;
  int wave = tid >> 5;
  int hi   = lane >> 4;     // lane half (K interleave / M offset)
  int nl   = lane & 15;     // row (A) / column (B,C) within tile
  int rowBase = (blockIdx.x * kWavesPerBlock + wave) * 16;
  int chunk   = blockIdx.y;
  int eStart  = chunk * kChunkSize;

  const unsigned ldsBase = (unsigned)(uintptr_t)(&smem[0][0][0]);

  // Resident A fragments. 16-bit A 16x32 layout:
  // lanes 0-15:  M=lane,    K = {0..7, 16..23}  (+32*kk)
  // lanes 16-31: M=lane-16, K = {8..15, 24..31} (+32*kk)
  v16bf a[16];
  const __bf16* xrow = xb + (size_t)(rowBase + nl) * kDim + (hi ? 8 : 0);
#pragma unroll
  for (int kk = 0; kk < 16; ++kk) {
    v8bf lo = *(const v8bf*)(xrow + kk * 32);
    v8bf hh = *(const v8bf*)(xrow + kk * 32 + 16);
    a[kk] = __builtin_shufflevector(lo, hh, 0, 1, 2, 3, 4, 5, 6, 7,
                                    8, 9, 10, 11, 12, 13, 14, 15);
  }

  float bestv[8];
  int   besti[8];
#pragma unroll
  for (int r = 0; r < 8; ++r) { bestv[r] = __builtin_inff(); besti[r] = 0x7fffffff; }

  // Stage tile 0: 16 rows x 1024 B = 1024 16-byte chunks over 256 threads.
  {
    int e0 = eStart;
#pragma unroll
    for (int i = 0; i < 1024 / kBlockThreads; ++i) {
      int c   = tid + kBlockThreads * i;
      int row = c >> 6;         // 64 16-byte chunks per row
      int c16 = c & 63;
      async_load_b128(eb + ((size_t)(e0 + row) << 9) + c16 * 8,
                      ldsBase + (unsigned)(row * (kLdsRow * 2) + c16 * 16));
    }
  }
  wait_asynccnt0();
  __syncthreads();

  const int bOff = hi * 8;    // B 32x16: lanes 0-15 K={0..7,16..23}, lanes 16-31 K={8..15,24..31}
  const int nTiles = kChunkSize / 16;   // 128 tiles per chunk
  for (int t = 0; t < nTiles; ++t) {
    int cur = t & 1;
    // Kick off async staging of the next tile before computing this one.
    if (t + 1 < nTiles) {
      int e1 = eStart + (t + 1) * 16;
      unsigned dstBase = ldsBase + (unsigned)((cur ^ 1) * kLdsTileB);
#pragma unroll
      for (int i = 0; i < 1024 / kBlockThreads; ++i) {
        int c   = tid + kBlockThreads * i;
        int row = c >> 6;
        int c16 = c & 63;
        async_load_b128(eb + ((size_t)(e1 + row) << 9) + c16 * 8,
                        dstBase + (unsigned)(row * (kLdsRow * 2) + c16 * 16));
      }
    }
    // Compute 16x16 dot tile from LDS-resident B with a 4-deep rotating
    // register pipeline for B fragments.
    v8f cacc = {0.f, 0.f, 0.f, 0.f, 0.f, 0.f, 0.f, 0.f};
    const __bf16* base = &smem[cur][nl][bOff];
    v8bf blo[4], bhi[4];
#pragma unroll
    for (int p = 0; p < 4; ++p) {
      blo[p] = *(const v8bf*)(base + p * 32);
      bhi[p] = *(const v8bf*)(base + p * 32 + 16);
    }
#pragma unroll
    for (int kk = 0; kk < 16; ++kk) {
      int cb = kk & 3;
      v16bf b = __builtin_shufflevector(blo[cb], bhi[cb], 0, 1, 2, 3, 4, 5, 6, 7,
                                        8, 9, 10, 11, 12, 13, 14, 15);
      cacc = __builtin_amdgcn_wmma_f32_16x16x32_bf16(
          false, a[kk], false, b, (short)0, cacc, false, false);
      if (kk + 4 < 16) {
        blo[cb] = *(const v8bf*)(base + (kk + 4) * 32);
        bhi[cb] = *(const v8bf*)(base + (kk + 4) * 32 + 16);
      }
    }
    // Scheduling pipeline: 8 DS reads ahead, then {1 WMMA, 2 DS reads} x12,
    // then the 4 drain WMMAs. (0x008 = MFMA/WMMA group, 0x100 = DS read)
    __builtin_amdgcn_sched_group_barrier(0x100, 8, 0);
#pragma unroll
    for (int g = 0; g < 12; ++g) {
      __builtin_amdgcn_sched_group_barrier(0x008, 1, 0);
      __builtin_amdgcn_sched_group_barrier(0x100, 2, 0);
    }
#pragma unroll
    for (int g = 0; g < 4; ++g) {
      __builtin_amdgcn_sched_group_barrier(0x008, 1, 0);
    }
    // C layout: VGPR r -> M = r (lanes 0-15) / 8+r (lanes 16-31); N = lane&15.
    int   e  = eStart + t * 16 + nl;
    float en = enorm[e];
#pragma unroll
    for (int r = 0; r < 8; ++r) {
      float score = __builtin_fmaf(-2.0f, cacc[r], en);  // ||E||^2 - 2*x.E
      if (score < bestv[r]) { bestv[r] = score; besti[r] = e; }
    }
    wait_asynccnt0();
    __syncthreads();
  }

  // Butterfly min+argmin across the 16 lanes sharing each output row
  // (xor masks < 16 stay inside each lane half); first-min tie-breaking.
#pragma unroll
  for (int r = 0; r < 8; ++r) {
    float v = bestv[r];
    int   i = besti[r];
#pragma unroll
    for (int m = 8; m >= 1; m >>= 1) {
      float ov = __shfl_xor(v, m, 32);
      int   oi = __shfl_xor(i, m, 32);
      if (ov < v || (ov == v && oi < i)) { v = ov; i = oi; }
    }
    if (nl == 0) {
      int row = rowBase + hi * 8 + r;
      pbestv[(size_t)chunk * kN + row] = v;
      pbesti[(size_t)chunk * kN + row] = i;
    }
  }
}

// ---------------------------------------------------------------------------
// Kernel 4: reduce per-chunk partial argmins -> final index per row
// ---------------------------------------------------------------------------
__global__ void argmin_reduce_kernel(const float* __restrict__ pbestv,
                                     const int* __restrict__ pbesti,
                                     int* __restrict__ ind) {
  int row = blockIdx.x * blockDim.x + threadIdx.x;
  if (row >= kN) return;
  float v = pbestv[row];
  int   i = pbesti[row];
#pragma unroll
  for (int c = 1; c < kChunks; ++c) {
    float ov = pbestv[(size_t)c * kN + row];
    int   oi = pbesti[(size_t)c * kN + row];
    if (ov < v || (ov == v && oi < i)) { v = ov; i = oi; }
  }
  ind[row] = i;
}

// ---------------------------------------------------------------------------
// Kernel 5: residual r = x - E[ind[n]] -> bf16
// ---------------------------------------------------------------------------
__global__ void resid_kernel(const float* __restrict__ x,
                             const float* __restrict__ E,
                             const int* __restrict__ ind,
                             __hip_bfloat16* __restrict__ rb) {
  int i = blockIdx.x * blockDim.x + threadIdx.x;
  if (i >= kN * kDim) return;
  int n = i >> 9;           // / kDim
  int k = i & (kDim - 1);
  float r = x[i] - E[(size_t)ind[n] * kDim + k];
  rb[i] = __float2bfloat16(r);
}

// ---------------------------------------------------------------------------
// Kernel 6: out = x + r @ W^T + b  (WMMA)
// out[m][j] = x[m][j] + pb[j] + sum_k r[m][k] * W[j][k]
// ---------------------------------------------------------------------------
__global__ __launch_bounds__(128) void proj_kernel(
    const __hip_bfloat16* __restrict__ rb_,
    const __hip_bfloat16* __restrict__ pwb_,
    const float* __restrict__ x,
    const float* __restrict__ pb,
    float* __restrict__ out) {
  const __bf16* rb = (const __bf16*)rb_;
  const __bf16* pw = (const __bf16*)pwb_;
  int lane = threadIdx.x & 31;
  int wave = threadIdx.x >> 5;
  int hi   = lane >> 4;
  int nl   = lane & 15;
  int rowBase = (blockIdx.x * 4 + wave) * 16;

  v16bf a[16];
  const __bf16* rrow = rb + (size_t)(rowBase + nl) * kDim + (hi ? 8 : 0);
#pragma unroll
  for (int kk = 0; kk < 16; ++kk) {
    v8bf lo = *(const v8bf*)(rrow + kk * 32);
    v8bf hh = *(const v8bf*)(rrow + kk * 32 + 16);
    a[kk] = __builtin_shufflevector(lo, hh, 0, 1, 2, 3, 4, 5, 6, 7,
                                    8, 9, 10, 11, 12, 13, 14, 15);
  }

  const int bOff = hi * 8;
  for (int j0 = 0; j0 < kDim; j0 += 16) {
    const __bf16* wrow = pw + (size_t)(j0 + nl) * kDim + bOff;
    v8f c = {0.f, 0.f, 0.f, 0.f, 0.f, 0.f, 0.f, 0.f};
#pragma unroll
    for (int kk = 0; kk < 16; ++kk) {
      v8bf lo = *(const v8bf*)(wrow + kk * 32);
      v8bf hh = *(const v8bf*)(wrow + kk * 32 + 16);
      v16bf b = __builtin_shufflevector(lo, hh, 0, 1, 2, 3, 4, 5, 6, 7,
                                        8, 9, 10, 11, 12, 13, 14, 15);
      c = __builtin_amdgcn_wmma_f32_16x16x32_bf16(
          false, a[kk], false, b, (short)0, c, false, false);
    }
    int   j    = j0 + nl;
    float bias = pb[j];
#pragma unroll
    for (int r = 0; r < 8; ++r) {
      int m = rowBase + hi * 8 + r;
      out[(size_t)m * kDim + j] = x[(size_t)m * kDim + j] + bias + c[r];
    }
  }
}

// ---------------------------------------------------------------------------
extern "C" void kernel_launch(void* const* d_in, const int* in_sizes, int n_in,
                              void* d_out, int out_size, void* d_ws, size_t ws_size,
                              hipStream_t stream) {
  const float* x  = (const float*)d_in[0];  // [8192, 512]
  const float* E  = (const float*)d_in[1];  // [8192, 512]
  const float* pw = (const float*)d_in[2];  // [512, 512]
  const float* pb = (const float*)d_in[3];  // [512]
  float* out = (float*)d_out;               // [8192, 512]

  // workspace layout (~25 MB)
  char* ws = (char*)d_ws;
  size_t off = 0;
  __hip_bfloat16* xb  = (__hip_bfloat16*)(ws + off); off += (size_t)kN * kDim * 2;
  __hip_bfloat16* eb  = (__hip_bfloat16*)(ws + off); off += (size_t)kEmb * kDim * 2;
  __hip_bfloat16* rbw = (__hip_bfloat16*)(ws + off); off += (size_t)kN * kDim * 2;
  __hip_bfloat16* pwb = (__hip_bfloat16*)(ws + off); off += (size_t)kDim * kDim * 2;
  float* enorm  = (float*)(ws + off); off += (size_t)kEmb * 4;
  float* pbestv = (float*)(ws + off); off += (size_t)kChunks * kN * 4;
  int*   pbesti = (int*)(ws + off);   off += (size_t)kChunks * kN * 4;
  int*   ind    = (int*)(ws + off);   off += (size_t)kN * 4;

  // 1) convert x and proj_w to bf16
  {
    int n = kN * kDim;
    cvt_bf16_kernel<<<(n + 255) / 256, 256, 0, stream>>>(x, xb, n);
    int m = kDim * kDim;
    cvt_bf16_kernel<<<(m + 255) / 256, 256, 0, stream>>>(pw, pwb, m);
  }
  // 2) codebook norms + bf16 convert (4 waves/block, one row per wave)
  enorm_cvt_kernel<<<kEmb / 4, 128, 0, stream>>>(E, eb, enorm);
  // 3) fused distance GEMM + partial argmin: 64 row-blocks x 4 codebook chunks
  {
    dim3 grid(kN / (16 * kWavesPerBlock), kChunks);
    dist_argmin_kernel<<<grid, kBlockThreads, 0, stream>>>(xb, eb, enorm, pbestv, pbesti);
  }
  // 4) reduce partial argmins
  argmin_reduce_kernel<<<(kN + 255) / 256, 256, 0, stream>>>(pbestv, pbesti, ind);
  // 5) residual
  {
    int n = kN * kDim;
    resid_kernel<<<(n + 255) / 256, 256, 0, stream>>>(x, E, ind, rbw);
  }
  // 6) projection + skip connection
  proj_kernel<<<kN / 64, 128, 0, stream>>>(rbw, pwb, x, pb, out);
}
